// LowerBlock_73065983639747
// MI455X (gfx1250) — 8-bit matrix operand order repaired and checked by reading, compiled, not run on hardware
//
#include <hip/hip_runtime.h>
#include <stdint.h>

#define BN_EPS 1e-5f

#define B_    64
#define C_    512
#define H_    28
#define Wd_   28
#define HW_   (H_ * Wd_)
#define P_    (B_ * HW_)
#define CHW_  (C_ * HW_)

typedef __attribute__((ext_vector_type(16))) int   v16i;
typedef __attribute__((ext_vector_type(8)))  float v8f;

__device__ __forceinline__ uint8_t fp8_sign(float v) {
    return v > 0.f ? (uint8_t)0x38 : (v < 0.f ? (uint8_t)0xB8 : (uint8_t)0x00);
}

__global__ void __launch_bounds__(256)
prep_w_kernel(const float* __restrict__ W,
              const float* __restrict__ bn_gamma, const float* __restrict__ bn_beta,
              const float* __restrict__ bn_mean,  const float* __restrict__ bn_var,
              uint8_t* __restrict__ Wfp8,
              float* __restrict__ kA, float* __restrict__ kB) {
    __shared__ float red[256];
    const int o = blockIdx.x;
    const int t = threadIdx.x;
    const float* row = W + (size_t)o * C_;
    float s = 0.f;
    for (int c = t; c < C_; c += 256) {
        float w = row[c];
        s += fabsf(w);
        Wfp8[(size_t)o * C_ + c] = fp8_sign(w);
    }
    red[t] = s;
    __syncthreads();
    for (int off = 128; off > 0; off >>= 1) {
        if (t < off) red[t] += red[t + off];
        __syncthreads();
    }
    if (t == 0) {
        float scale = red[0] * (1.0f / (float)C_);
        float rs = rsqrtf(bn_var[o] + BN_EPS);
        float g  = bn_gamma[o] * rs;
        kA[o] = g * scale;
        kB[o] = bn_beta[o] - g * bn_mean[o];
    }
}

#define TPAD 68

__global__ void __launch_bounds__(256)
binarize_x_kernel(const float* __restrict__ x,
                  const float* __restrict__ rsign_bias,
                  uint8_t* __restrict__ A8) {
    __shared__ uint8_t tile[64 * TPAD];
    const int c0 = blockIdx.x * 64;
    const int p0 = blockIdx.y * 64;
    const int t  = threadIdx.x;

    for (int i = t; i < 64 * 64; i += 256) {
        int r   = i >> 6;
        int col = i & 63;
        int c = c0 + r;
        int p = p0 + col;
        int b  = p / HW_;
        int hw = p - b * HW_;
        float v = x[(size_t)b * CHW_ + (size_t)c * HW_ + hw] + rsign_bias[c];
        tile[r * TPAD + col] = fp8_sign(v);
    }
    __syncthreads();

    for (int i = t; i < 64 * 16; i += 256) {
        int pr = i >> 4;
        int c4 = i & 15;
        uint32_t w;
        w  = (uint32_t)tile[(4 * c4 + 0) * TPAD + pr];
        w |= (uint32_t)tile[(4 * c4 + 1) * TPAD + pr] << 8;
        w |= (uint32_t)tile[(4 * c4 + 2) * TPAD + pr] << 16;
        w |= (uint32_t)tile[(4 * c4 + 3) * TPAD + pr] << 24;
        *(uint32_t*)(A8 + (size_t)(p0 + pr) * C_ + (c0 + 4 * c4)) = w;
    }
}

#define BROW 528
#define SM_BYTES (8 * 64 * 17 * 4)

__global__ void __launch_bounds__(256)
bgemm_epilogue_kernel(const uint8_t* __restrict__ A8,
                      const uint8_t* __restrict__ Wfp8,
                      const float* __restrict__ kA, const float* __restrict__ kB,
                      const float* __restrict__ x,
                      const float* __restrict__ pr_slope,
                      const float* __restrict__ pr_shift,
                      const float* __restrict__ pr_bias,
                      float* __restrict__ out) {
    __shared__ uint8_t smem[SM_BYTES];
    const int p0   = blockIdx.x * 128;
    const int o0   = blockIdx.y * 64;
    const int t    = threadIdx.x;
    const int wid  = t >> 5;
    const int lane = t & 31;
    const int half = lane >> 4;
    const int l15  = lane & 15;

    for (int i = t; i < 64 * 32; i += 256) {
        int r  = i >> 5;
        int cb = (i & 31) * 16;
        uint4 d = *(const uint4*)(Wfp8 + (size_t)(o0 + r) * C_ + cb);
        *(uint4*)(smem + r * BROW + cb) = d;
    }
    __syncthreads();

    const int pw = p0 + wid * 16;

    v8f acc[4] = {};
    const uint8_t* arow = A8 + (size_t)(pw + l15) * C_ + (half ? 16 : 0);

    for (int k0 = 0; k0 < C_; k0 += 128) {
        v16i a;
#pragma unroll
        for (int j = 0; j < 8; ++j) {
            uint2 d = *(const uint2*)(arow + k0 + 32 * (j >> 1) + 8 * (j & 1));
            a[2 * j]     = (int)d.x;
            a[2 * j + 1] = (int)d.y;
        }
#pragma unroll
        for (int tt = 0; tt < 4; ++tt) {
            const uint8_t* brow = smem + (tt * 16 + l15) * BROW + k0 + (half ? 16 : 0);
            v16i b;
#pragma unroll
            for (int j = 0; j < 4; ++j) {
                uint4 d = *(const uint4*)(brow + 32 * j);
                b[4 * j]     = (int)d.x;
                b[4 * j + 1] = (int)d.y;
                b[4 * j + 2] = (int)d.z;
                b[4 * j + 3] = (int)d.w;
            }
            acc[tt] = __builtin_amdgcn_wmma_f32_16x16x128_fp8_fp8(
                a, b, (short)0, acc[tt], false, false);
        }
    }

    __syncthreads();

    float* accLds = (float*)smem + wid * (64 * 17);
#pragma unroll
    for (int tt = 0; tt < 4; ++tt) {
#pragma unroll
        for (int r = 0; r < 8; ++r) {
            int o_local = tt * 16 + l15;
            int p_local = r + 8 * half;
            accLds[o_local * 17 + p_local] = acc[tt][r];
        }
    }

    const int p  = pw + l15;
    const int b  = p / HW_;
    const int hw = p - b * HW_;
    for (int it = 0; it < 32; ++it) {
        int o_local = 2 * it + half;
        int o = o0 + o_local;
        float accv = accLds[o_local * 17 + l15];
        float a_o = kA[o];
        float b_o = kB[o];
        float sl  = pr_slope[o];
        float sh  = pr_shift[o];
        float bi  = pr_bias[o];
        size_t idx = (size_t)b * CHW_ + (size_t)o * HW_ + hw;
        float s  = fmaf(a_o, accv, b_o) + x[idx];
        float tv = s - sh;
        out[idx] = (tv > 0.f ? tv : sl * tv) + bi;
    }
}

extern "C" void kernel_launch(void* const* d_in, const int* in_sizes, int n_in,
                              void* d_out, int out_size, void* d_ws, size_t ws_size,
                              hipStream_t stream) {
    (void)in_sizes; (void)n_in; (void)out_size; (void)ws_size;
    const float* x    = (const float*)d_in[0];
    const float* rsb  = (const float*)d_in[1];
    const float* W    = (const float*)d_in[2];
    const float* g    = (const float*)d_in[3];
    const float* be   = (const float*)d_in[4];
    const float* mu   = (const float*)d_in[5];
    const float* var  = (const float*)d_in[6];
    const float* sl   = (const float*)d_in[7];
    const float* sh   = (const float*)d_in[8];
    const float* bi   = (const float*)d_in[9];
    float* out = (float*)d_out;

    uint8_t* ws   = (uint8_t*)d_ws;
    uint8_t* A8   = ws;
    uint8_t* Wfp8 = ws + (size_t)P_ * C_;
    float*   kA   = (float*)(ws + (size_t)P_ * C_ + (size_t)C_ * C_);
    float*   kB   = kA + C_;

    prep_w_kernel<<<C_, 256, 0, stream>>>(W, g, be, mu, var, Wfp8, kA, kB);
    binarize_x_kernel<<<dim3(C_ / 64, P_ / 64), 256, 0, stream>>>(x, rsb, A8);
    bgemm_epilogue_kernel<<<dim3(P_ / 128, C_ / 64), 256, 0, stream>>>(
        A8, Wfp8, kA, kB, x, sl, sh, bi, out);
}
